// Memory_90031104459201
// MI455X (gfx1250) — compile-verified, matrix operand order checked by reading, not run on hardware
//
#include <hip/hip_runtime.h>
#include <math.h>

#define NUM_CLS  1000
#define NUM_SRC  1000
#define FEAT_DIM 256
#define NTOT     65536
#define NCOL     (NUM_CLS + NUM_SRC)   // 2000 = 125 * 16 exactly
#define NTILES   (NCOL / 16)           // 125
#define EPSN     1e-12f
#define BM       64
#define LDA      260                   // A tile row stride (floats); 260 % 64 = 4
#define BROWB    528                   // bf16 B row stride in BYTES (264 bf16), 16B aligned
#define BTILEB   (32 * BROWB)          // one H or L tile-pair plane: 16896 B
#define BBUFB    (2 * BTILEB)          // H+L planes per buffer: 33792 B

typedef __attribute__((ext_vector_type(8)))  float v8f;
typedef __attribute__((ext_vector_type(16))) __bf16 v16bf;
typedef __attribute__((ext_vector_type(8)))  unsigned short v8us;
typedef __attribute__((ext_vector_type(16))) unsigned short v16us;

__device__ __forceinline__ float wave_reduce_sum(float x) {
#pragma unroll
  for (int m = 16; m >= 1; m >>= 1) x += __shfl_xor(x, m, 32);
  return x;
}

// RNE float -> bf16 (bit exact, no __bf16 arithmetic needed)
__device__ __forceinline__ unsigned short f2bf(float x) {
  unsigned u = __float_as_uint(x);
  unsigned r = u + 0x7fffu + ((u >> 16) & 1u);
  return (unsigned short)(r >> 16);
}
__device__ __forceinline__ float bf2f(unsigned short h) {
  return __uint_as_float(((unsigned)h) << 16);
}

// ---- CDNA5 async global->LDS copy (ASYNCcnt-tracked) ----
__device__ __forceinline__ void async_load_b128(const void* gaddr, void* lds) {
  unsigned loff = (unsigned)(unsigned long long)lds;  // low 32b of LDS generic ptr = LDS offset
  asm volatile("global_load_async_to_lds_b128 %0, %1, off"
               :: "v"(loff), "v"(gaddr) : "memory");
}
__device__ __forceinline__ void wait_asynccnt_le8() {
  asm volatile("s_wait_asynccnt 0x8" ::: "memory");
}
__device__ __forceinline__ void wait_asynccnt_0() {
  asm volatile("s_wait_asynccnt 0x0" ::: "memory");
}

// ---------------- Kernel 1: row L2-norm + segment sum by label ----------------
__global__ __launch_bounds__(256) void k_norm_segsum(
    const float* __restrict__ feat, const int* __restrict__ label,
    float* __restrict__ sums, float* __restrict__ counts) {
  const int wave = threadIdx.x >> 5;
  const int lane = threadIdx.x & 31;
  const int row  = blockIdx.x * 8 + wave;
  const float* fr = feat + (size_t)row * FEAT_DIM;
  float v[8]; float ss = 0.f;
#pragma unroll
  for (int i = 0; i < 8; ++i) { v[i] = fr[lane + i * 32]; ss += v[i] * v[i]; }
  ss = wave_reduce_sum(ss);
  const float inv = 1.f / fmaxf(sqrtf(ss), EPSN);
  const int lab = label[row];
  float* sr = sums + (size_t)lab * FEAT_DIM;
#pragma unroll
  for (int i = 0; i < 8; ++i) atomicAdd(&sr[lane + i * 32], v[i] * inv);
  if (lane == 0) atomicAdd(&counts[lab], 1.f);
}

// ------- Kernel 2: batch_center -> new_memory; emit bf16 hi/lo memo planes -------
__global__ __launch_bounds__(256) void k_build_memo(
    const float* __restrict__ sums, const float* __restrict__ counts,
    const float* __restrict__ memory, const float* __restrict__ source_memo,
    unsigned short* __restrict__ memoH, unsigned short* __restrict__ memoL) {
  const int wave = threadIdx.x >> 5;
  const int lane = threadIdx.x & 31;
  const int row  = blockIdx.x * 8 + wave;
  if (row >= NCOL) return;
  float m[8];
  if (row >= NUM_CLS) {  // source_memo rows pass through
    const float* s = source_memo + (size_t)(row - NUM_CLS) * FEAT_DIM;
#pragma unroll
    for (int i = 0; i < 8; ++i) m[i] = s[lane + i * 32];
  } else {
    const float* sr = sums + (size_t)row * FEAT_DIM;
    const float* mr = memory + (size_t)row * FEAT_DIM;
    float s[8]; float ss = 0.f;
#pragma unroll
    for (int i = 0; i < 8; ++i) {
      s[i] = sr[lane + i * 32]; m[i] = mr[lane + i * 32]; ss += s[i] * s[i];
    }
    ss = wave_reduce_sum(ss);
    const float flag = (counts[row] > 0.f) ? 1.f : 0.f;
    const float invs = 1.f / fmaxf(sqrtf(ss), EPSN);
    float dotp = 0.f;
#pragma unroll
    for (int i = 0; i < 8; ++i) { s[i] = s[i] * invs * flag; dotp += m[i] * s[i]; }
    dotp = wave_reduce_sum(dotp);
    const float w = 1.f - (1.f - dotp) * flag;   // empty class -> keep old memory
    float ss2 = 0.f;
#pragma unroll
    for (int i = 0; i < 8; ++i) { m[i] = w * m[i] + (1.f - w) * s[i]; ss2 += m[i] * m[i]; }
    ss2 = wave_reduce_sum(ss2);
    const float inv2 = 1.f / fmaxf(sqrtf(ss2), EPSN);
#pragma unroll
    for (int i = 0; i < 8; ++i) m[i] *= inv2;
  }
  unsigned short* dh = memoH + (size_t)row * FEAT_DIM;
  unsigned short* dl = memoL + (size_t)row * FEAT_DIM;
#pragma unroll
  for (int i = 0; i < 8; ++i) {
    const unsigned short hh = f2bf(m[i]);
    dh[lane + i * 32] = hh;
    dl[lane + i * 32] = f2bf(m[i] - bf2f(hh));
  }
}

// ---- Kernel 3: bf16x3-split fp32 GEMM on v_wmma_f32_16x16x32_bf16 + fused NLL ----
__global__ __launch_bounds__(256) void k_logits_loss(
    const float* __restrict__ feat, const int* __restrict__ label,
    const unsigned short* __restrict__ memoH, const unsigned short* __restrict__ memoL,
    float* __restrict__ out) {
  // Union: A tile (64x260 f32 = 66560 B) then double-buffered bf16 B planes (67584 B)
  __shared__ __align__(16) unsigned char sbuf[BBUFB * 2];   // 67584 B
  __shared__ float sSum[BM];
  __shared__ float sLab[BM];
  float* As = (float*)sbuf;

  const int t    = threadIdx.x;
  const int wave = t >> 5;
  const int lane = t & 31;
  const int rowBase = blockIdx.x * BM;

  // ---- Stage 64x256 feat tile into LDS ----
  const float4* fg = (const float4*)(feat + (size_t)rowBase * FEAT_DIM);
#pragma unroll
  for (int i = 0; i < 16; ++i) {
    const int flat = i * 256 + t;
    const int r = flat >> 6;
    const int c4 = flat & 63;
    const float4 x = fg[flat];
    float* dst = &As[r * LDA + c4 * 4];
    dst[0] = x.x; dst[1] = x.y; dst[2] = x.z; dst[3] = x.w;
  }
  if (t < BM) { sSum[t] = 0.f; sLab[t] = 0.f; }
  __syncthreads();

  // ---- Normalize 8 rows per wave in LDS ----
  {
    const int r0 = wave * 8;
#pragma unroll
    for (int r = 0; r < 8; ++r) {
      float* Ar = &As[(r0 + r) * LDA];
      float ss = 0.f;
#pragma unroll
      for (int i = 0; i < 8; ++i) { const float x = Ar[lane + i * 32]; ss += x * x; }
      ss = wave_reduce_sum(ss);
      const float inv = 1.f / fmaxf(sqrtf(ss), EPSN);
#pragma unroll
      for (int i = 0; i < 8; ++i) Ar[lane + i * 32] *= inv;
    }
  }
  __syncthreads();

  const int mt   = wave & 3;      // M-tile (16 rows)
  const int half = wave >> 2;     // which tile of the staged pair
  const int h    = lane >> 4;     // lane half (ISA 16-bit A/B K-group select)
  const int ln   = lane & 15;
  const int mrowL = mt * 16 + h * 8;

  // ---- Split this wave's A rows into bf16 hi/lo fragments in registers ----
  // 16-bit A 16x32 layout: lane half h, element j<8 -> K = g*32 + h*8 + j,
  //                        element j>=8 -> K = g*32 + h*8 + 16 + (j-8)
  v16bf aH[8], aL[8];
#pragma unroll
  for (int g = 0; g < 8; ++g) {
    const float* Ap = &As[(mt * 16 + ln) * LDA + g * 32 + h * 8];
    v16us Hu, Lu;
#pragma unroll
    for (int j = 0; j < 8; ++j) {
      const float x0 = Ap[j];
      const unsigned short h0 = f2bf(x0);
      Hu[j] = h0; Lu[j] = f2bf(x0 - bf2f(h0));
      const float x1 = Ap[16 + j];
      const unsigned short h1 = f2bf(x1);
      Hu[8 + j] = h1; Lu[8 + j] = f2bf(x1 - bf2f(h1));
    }
    aH[g] = __builtin_bit_cast(v16bf, Hu);
    aL[g] = __builtin_bit_cast(v16bf, Lu);
  }

  int labV[8];
#pragma unroll
  for (int r = 0; r < 8; ++r) labV[r] = label[rowBase + mrowL + r];
  float accSum[8];
#pragma unroll
  for (int r = 0; r < 8; ++r) accSum[r] = 0.f;

  __syncthreads();   // A consumed; sbuf now free for B planes

  // ---- Async staging of a tile pair (32 rows) of H and L bf16 planes ----
  auto stage_pair = [&](int buf, int ntBase) {
#pragma unroll
    for (int i = 0; i < 8; ++i) {
      const int u = t + i * 256;          // 2048 x 16B units (1024 H + 1024 L)
      const int arr = u >> 10;            // 0 = H, 1 = L
      const int rem = u & 1023;
      const int rp = rem >> 5;            // row in pair (32 rows, 32 units each)
      const int c16 = rem & 31;           // 16B unit within row
      int grow = ntBase * 16 + rp;
      grow = grow < NCOL ? grow : NCOL - 1;
      const unsigned short* src = (arr == 0 ? memoH : memoL);
      const void* g = (const unsigned char*)(src + (size_t)grow * FEAT_DIM) + c16 * 16;
      void* l = sbuf + buf * BBUFB + arr * BTILEB + rp * BROWB + c16 * 16;
      async_load_b128(g, l);
    }
  };

  stage_pair(0, 0);
  int buf = 0;
  for (int it = 0; it < 63; ++it) {          // 63 pairs cover 125 tiles (+1 pad)
    if (it < 62) { stage_pair(buf ^ 1, 2 * (it + 1)); wait_asynccnt_le8(); }
    else         { wait_asynccnt_0(); }
    __syncthreads();                          // B(buf) visible to all waves

    const int nt = 2 * it + half;
    if (__builtin_amdgcn_readfirstlane(nt) < NTILES) {  // scalar branch: EXEC untouched
      const unsigned char* Hrow = sbuf + buf * BBUFB + (half * 16 + ln) * BROWB + h * 16;
      const unsigned char* Lrow = Hrow + BTILEB;
      v8f c0 = {}, c1 = {};
#pragma unroll
      for (int g = 0; g < 8; ++g) {
        const v8us bh0 = *(const v8us*)(Hrow + g * 64);
        const v8us bh1 = *(const v8us*)(Hrow + g * 64 + 32);
        const v8us bl0 = *(const v8us*)(Lrow + g * 64);
        const v8us bl1 = *(const v8us*)(Lrow + g * 64 + 32);
        const v16bf bH = __builtin_bit_cast(v16bf,
            __builtin_shufflevector(bh0, bh1, 0,1,2,3,4,5,6,7,8,9,10,11,12,13,14,15));
        const v16bf bL = __builtin_bit_cast(v16bf,
            __builtin_shufflevector(bl0, bl1, 0,1,2,3,4,5,6,7,8,9,10,11,12,13,14,15));
        v8f& c = (g < 4) ? c0 : c1;          // two independent accumulation chains
        c = __builtin_amdgcn_wmma_f32_16x16x32_bf16(false, aH[g], false, bH,
                                                    (short)0, c, false, false);
        c = __builtin_amdgcn_wmma_f32_16x16x32_bf16(false, aH[g], false, bL,
                                                    (short)0, c, false, false);
        c = __builtin_amdgcn_wmma_f32_16x16x32_bf16(false, aL[g], false, bH,
                                                    (short)0, c, false, false);
      }
      const int colG = nt * 16 + ln;
#pragma unroll
      for (int r = 0; r < 8; ++r) {
        const float x = c0[r] + c1[r];
        accSum[r] += expf(x - 1.f);          // unit vectors: logits <= 1
        if (labV[r] == colG) sLab[mrowL + r] = x;
      }
    }
    __syncthreads();                          // done reading B(buf)
    buf ^= 1;
  }

  // ---- Reduce per-row partials across the 16-lane group; merge waves in LDS ----
#pragma unroll
  for (int r = 0; r < 8; ++r) {
    float s = accSum[r];
    s += __shfl_xor(s, 1, 32);
    s += __shfl_xor(s, 2, 32);
    s += __shfl_xor(s, 4, 32);
    s += __shfl_xor(s, 8, 32);
    if (ln == 0) atomicAdd(&sSum[mrowL + r], s);
  }
  __syncthreads();

  if (t < BM) {
    const float loss = -(sLab[t] - 1.f - logf(sSum[t]));
    atomicAdd(out, loss * (1.f / (float)NTOT));
  }
}

// -------------------------------- launcher --------------------------------
extern "C" void kernel_launch(void* const* d_in, const int* in_sizes, int n_in,
                              void* d_out, int out_size, void* d_ws, size_t ws_size,
                              hipStream_t stream) {
  const float* feat        = (const float*)d_in[0];
  const int*   label       = (const int*)d_in[1];
  const float* memory      = (const float*)d_in[2];
  const float* source_memo = (const float*)d_in[3];

  float* sums   = (float*)d_ws;                                   // [1000*256] f32
  float* counts = sums + NUM_CLS * FEAT_DIM;                      // [1000] f32
  unsigned short* memoH = (unsigned short*)(counts + NUM_CLS);    // [2000*256] bf16 bits
  unsigned short* memoL = memoH + (size_t)NCOL * FEAT_DIM;        // [2000*256] bf16 bits

  hipMemsetAsync(d_ws, 0, (size_t)(NUM_CLS * FEAT_DIM + NUM_CLS) * sizeof(float), stream);
  hipMemsetAsync(d_out, 0, sizeof(float), stream);

  k_norm_segsum<<<NTOT / 8, 256, 0, stream>>>(feat, label, sums, counts);
  k_build_memo<<<(NCOL + 7) / 8, 256, 0, stream>>>(sums, counts, memory, source_memo,
                                                   memoH, memoL);
  k_logits_loss<<<NTOT / BM, 256, 0, stream>>>(feat, label, memoH, memoL, (float*)d_out);
}
